// Block_28613072126771
// MI455X (gfx1250) — compile-verified, matrix operand order checked by reading, not run on hardware
//
#include <hip/hip_runtime.h>
#include <hip/hip_bf16.h>
#include <stdint.h>

// ---------------------------------------------------------------------------
// Transformer block (attn + MLP, pre-LN, causal) for MI455X / gfx1250.
// All GEMMs via v_wmma_f32_16x16x32_f16 (wave32 WMMA, f32 accumulate).
// Fragments are K-major -> 2x b128 loads each; GEMM B operand is staged
// through LDS via the Tensor Data Mover (TDM) and reused by all 8 waves.
// ---------------------------------------------------------------------------

typedef _Float16 h16_t;
typedef __attribute__((ext_vector_type(16))) _Float16 v16h;
typedef __attribute__((ext_vector_type(8)))  _Float16 v8h;
typedef __attribute__((ext_vector_type(8)))  float    v8f;
typedef __attribute__((ext_vector_type(4)))  unsigned int u32x4;
typedef __attribute__((ext_vector_type(8)))  int          i32x8;
typedef __attribute__((ext_vector_type(4)))  int          i32x4;

#define N_EMBD   384
#define N_HEADS  6
#define HEAD_SZ  64
#define D_FF     1536
#define SEQ_T    256
#define BATCH_B  64
#define BT       (BATCH_B * SEQ_T)        // 16384
#define QKV_LD   (3 * N_EMBD)             // 1152
#define KCHUNK   384                      // B-staging chunk (divides 384, 1536)

#if defined(__has_builtin)
#if __has_builtin(__builtin_amdgcn_tensor_load_to_lds) && \
    __has_builtin(__builtin_amdgcn_s_wait_tensorcnt)
#define HAVE_TDM 1
#endif
#endif

// ---- WMMA helpers ----------------------------------------------------------

__device__ __forceinline__ v8f wmma_f16(v16h a, v16h b, v8f c) {
  // D = A(16x32 f16) * B(32x16 f16) + C(16x16 f32)
  return __builtin_amdgcn_wmma_f32_16x16x32_f16(
      /*neg_a=*/false, a, /*neg_b=*/false, b,
      /*c_mod=*/(short)0, c, /*reuse_a=*/false, /*reuse_b=*/false);
}

// K-major fragment loader (A-frags, and B-frags stored transposed).
// ISA wave32 layout: lane L -> row/col = L&15;
// elems 0..7 -> K = (L>>4)*8 + i, elems 8..15 -> K = 16 + (L>>4)*8 + i.
// Per lane: two contiguous 16-byte chunks -> two b128 loads (global or ds).
__device__ __forceinline__ v16h load_kfrag(const h16_t* p, int ld, int lane) {
  const h16_t* r = p + (size_t)(lane & 15) * ld + ((lane >> 4) << 3);
  v8h lo = *(const v8h*)r;
  v8h hi = *(const v8h*)(r + 16);
  return __builtin_shufflevector(lo, hi, 0, 1, 2, 3, 4, 5, 6, 7,
                                 8, 9, 10, 11, 12, 13, 14, 15);
}

// ---- Weight packing / conversion (all outputs TRANSPOSED = K-major) --------

// Wq/Wk/Wv [H][C][HS] -> Wqkv_t [1152][384] : Wqkv_t[s*384+h*64+k][c]
__global__ void pack_qkv_kernel(const float* __restrict__ Wq,
                                const float* __restrict__ Wk,
                                const float* __restrict__ Wv,
                                h16_t* __restrict__ outT) {
  int idx = blockIdx.x * 256 + threadIdx.x;
  if (idx >= N_EMBD * QKV_LD) return;
  int n = idx / N_EMBD;           // 0..1151
  int c = idx % N_EMBD;
  int s = n / N_EMBD;
  int h = (n % N_EMBD) / HEAD_SZ;
  int k = n % HEAD_SZ;
  const float* W = (s == 0) ? Wq : (s == 1) ? Wk : Wv;
  outT[idx] = (h16_t)W[((size_t)h * N_EMBD + c) * HEAD_SZ + k];
}

// in [K][N] f32 -> outT [N][K] f16
__global__ void cvt_t_kernel(const float* __restrict__ in,
                             h16_t* __restrict__ outT, int K, int N) {
  int idx = blockIdx.x * 256 + threadIdx.x;
  if (idx >= K * N) return;
  int n = idx / K;
  int k = idx % K;
  outT[idx] = (h16_t)in[(size_t)k * N + n];
}

// ---- V transpose: qkv [BT][1152] -> Vt [B*H][HS][T] ------------------------

__global__ void vtrans_kernel(const h16_t* __restrict__ qkv,
                              h16_t* __restrict__ Vt) {
  __shared__ h16_t tile[HEAD_SZ][HEAD_SZ + 1];
  const int bh = blockIdx.x >> 2;          // b*6 + h
  const int tt = blockIdx.x & 3;           // 64-wide t tile
  const int b  = bh / N_HEADS, h = bh % N_HEADS;
  const h16_t* V = qkv + (size_t)b * SEQ_T * QKV_LD + 2 * N_EMBD + h * HEAD_SZ;
  const int tx = threadIdx.x & 63;
  const int ty = threadIdx.x >> 6;         // 0..3
#pragma unroll
  for (int i = 0; i < 16; ++i) {
    int t = ty * 16 + i;                   // local t
    tile[tx][t] = V[(size_t)(tt * 64 + t) * QKV_LD + tx];  // tile[hs][t]
  }
  __syncthreads();
  h16_t* O = Vt + (size_t)bh * HEAD_SZ * SEQ_T + tt * 64;
#pragma unroll
  for (int i = 0; i < 16; ++i) {
    int hs = ty * 16 + i;
    O[(size_t)hs * SEQ_T + tx] = tile[hs][tx];
  }
}

// ---- LayerNorm (one wave per row, C=384) -----------------------------------

__global__ void ln_kernel(const float* __restrict__ x,
                          const float* __restrict__ g,
                          const float* __restrict__ beta,
                          h16_t* __restrict__ out) {
  const int row  = blockIdx.x * 8 + (threadIdx.x >> 5);
  const int lane = threadIdx.x & 31;
  const float* xr = x + (size_t)row * N_EMBD;
  float s = 0.f, ss = 0.f;
#pragma unroll
  for (int c = lane; c < N_EMBD; c += 32) {
    float v = xr[c];
    s += v; ss += v * v;
  }
#pragma unroll
  for (int off = 16; off; off >>= 1) {
    s  += __shfl_xor(s,  off, 32);
    ss += __shfl_xor(ss, off, 32);
  }
  const float mu   = s * (1.f / N_EMBD);
  const float var  = ss * (1.f / N_EMBD) - mu * mu;
  const float rstd = rsqrtf(var + 1e-5f);
  h16_t* orow = out + (size_t)row * N_EMBD;
#pragma unroll
  for (int c = lane; c < N_EMBD; c += 32)
    orow[c] = (h16_t)((xr[c] - mu) * rstd * g[c] + beta[c]);
}

// ---- WMMA GEMM: C[M,N] = A[M,K] * Bt[N,K]^T --------------------------------
// Block = 8 waves = 256x64 output tile.  The 64-col B block is DMA'd into LDS
// by the Tensor Data Mover (one 2-D descriptor per 48 KB K-chunk, issued by
// wave 0, tracked with TENSORcnt) and reused by all waves.  Each wave:
// 32x64 strip = 2 A-frags (global b128) + 4 B-frags (ds b128) -> 8 WMMAs
// per 32-K step.

__global__ void gemm16_kernel(const h16_t* __restrict__ A,
                              const h16_t* __restrict__ Bt,
                              const float* __restrict__ bias,
                              const float* __restrict__ resid,
                              float* __restrict__ outf,
                              h16_t* __restrict__ outh,
                              int M, int N, int K, int relu) {
  __shared__ h16_t Bs[64 * KCHUNK];        // 48 KB
  const int lane = threadIdx.x & 31;
  const int wave = threadIdx.x >> 5;
  const int n64  = N >> 6;
  const int bm   = blockIdx.x / n64;
  const int bn   = blockIdx.x % n64;
  const int rowbase = bm * 256 + wave * 32;

  const h16_t* Ab  = A  + (size_t)rowbase * K;
  const h16_t* BtB = Bt + (size_t)bn * 64 * K;

  v8f acc[2][4] = {};
  for (int k0 = 0; k0 < K; k0 += KCHUNK) {
#if HAVE_TDM
    // Stage Bt[bn*64 .. +64][k0 .. k0+384] into LDS with one TDM descriptor:
    // 2-D tile, data_size=2B, tile = 384 x 64, row stride = K, packed in LDS.
    if (wave == 0) {
      const uint64_t ga  = (uint64_t)(uintptr_t)(BtB + k0);   // tile start
      const unsigned  lds = (unsigned)(uintptr_t)(&Bs[0]);    // LDS byte addr
      // D# group0: count=1 | lds_addr | global_addr[56:0] | type=2 (bits 127:126)
      u32x4 g0 = { 1u, lds, (unsigned)ga,
                   (unsigned)(((ga >> 32) & 0x1FFFFFFu) | 0x80000000u) };
      // D# group1: data_size=1(2B); tensor_dim0=384; tensor_dim1=64;
      // tile_dim0=384; tile_dim1=64; tensor_dim0_stride=K.
      i32x8 g1 = { 0x00010000, (int)(KCHUNK << 16), (int)(64 << 16),
                   (int)(KCHUNK << 16), 64, K, 0, 0 };
      i32x4 gz = { 0, 0, 0, 0 };
#if __clang_major__ >= 23
      i32x8 gz8 = { 0, 0, 0, 0, 0, 0, 0, 0 };
      __builtin_amdgcn_tensor_load_to_lds(g0, g1, gz, gz, gz8, 0);
#else
      __builtin_amdgcn_tensor_load_to_lds(g0, g1, gz, gz, 0);
#endif
      __builtin_amdgcn_s_wait_tensorcnt(0);
    }
#else
    for (int i = threadIdx.x * 8; i < 64 * KCHUNK; i += 256 * 8) {
      const int row = i / KCHUNK, kk = i % KCHUNK;
      *(v8h*)(Bs + i) = *(const v8h*)(BtB + (size_t)row * K + k0 + kk);
    }
#endif
    if (k0 + KCHUNK < K) {                 // hint the next chunks into cache
      __builtin_prefetch(Ab + k0 + KCHUNK, 0, 0);
      __builtin_prefetch(BtB + k0 + KCHUNK, 0, 0);
    }
    __syncthreads();

#pragma unroll 2
    for (int kk = 0; kk < KCHUNK; kk += 32) {
      v16h a0 = load_kfrag(Ab + k0 + kk,                  K, lane);
      v16h a1 = load_kfrag(Ab + (size_t)16 * K + k0 + kk, K, lane);
#pragma unroll
      for (int ct = 0; ct < 4; ++ct) {
        v16h b = load_kfrag(Bs + ct * 16 * KCHUNK + kk, KCHUNK, lane);
        acc[0][ct] = wmma_f16(a0, b, acc[0][ct]);
        acc[1][ct] = wmma_f16(a1, b, acc[1][ct]);
      }
    }
    __syncthreads();
  }

  const int col = lane & 15;
  const int rb  = (lane >> 4) * 8;
#pragma unroll
  for (int mi = 0; mi < 2; ++mi) {
#pragma unroll
    for (int ct = 0; ct < 4; ++ct) {
      const int n = bn * 64 + ct * 16 + col;
      const float bv = bias ? bias[n] : 0.0f;
#pragma unroll
      for (int r = 0; r < 8; ++r) {
        const int m = rowbase + mi * 16 + rb + r;
        const size_t idx = (size_t)m * N + n;
        float v = acc[mi][ct][r] + bv;
        if (relu)  v = fmaxf(v, 0.0f);
        if (resid) v += resid[idx];
        if (outh)  outh[idx] = (h16_t)v;
        else       outf[idx] = v;
      }
    }
  }
}

// ---- Fused causal attention ------------------------------------------------
// qkv: [B*T, 1152] f16 (Q|K per head, row-major), Vt: [B*H][HS][T] f16.
// One block = 2 waves, each wave owns a 16-row query tile. Per-wave LDS:
// 16x256 f32 scores + 16x256 f16 probabilities (48 KB / block; WGP has 320 KB).

__global__ void attn_kernel(const h16_t* __restrict__ qkv,
                            const h16_t* __restrict__ Vt,
                            h16_t* __restrict__ attn_out) {
  extern __shared__ float smem[];
  const int lane = threadIdx.x & 31;
  const int wave = threadIdx.x >> 5;           // 0..1
  const int qblk = blockIdx.x & 7;             // 8 blocks per (b,h)
  const int h    = (blockIdx.x >> 3) % N_HEADS;
  const int b    = (blockIdx.x >> 3) / N_HEADS;
  const int qbase = qblk * 32 + wave * 16;

  const h16_t* Qp  = qkv + (size_t)b * SEQ_T * QKV_LD + h * HEAD_SZ;
  const h16_t* Kp  = Qp + N_EMBD;
  const h16_t* Vtb = Vt + (size_t)(b * N_HEADS + h) * HEAD_SZ * SEQ_T;

  float* S = smem + (size_t)wave * 16 * SEQ_T;                            // 16x256 f32
  h16_t* P = (h16_t*)(smem + 2 * 16 * SEQ_T) + (size_t)wave * 16 * SEQ_T; // 16x256 f16
  const float scale = 0.125f;                   // 1/sqrt(64)

  // Phase 1: S = scale * Q K^T, causal-masked, into LDS.
  for (int jt = 0; jt < SEQ_T / 16; ++jt) {
    v8f acc = {};
#pragma unroll
    for (int kb = 0; kb < 2; ++kb) {            // HS=64 -> two K=32 steps
      v16h a  = load_kfrag(Qp + (size_t)qbase     * QKV_LD + kb * 32, QKV_LD, lane);
      v16h bb = load_kfrag(Kp + (size_t)(jt * 16) * QKV_LD + kb * 32, QKV_LD, lane);
      acc = wmma_f16(a, bb, acc);
    }
    const int col = lane & 15, rb = (lane >> 4) * 8;
#pragma unroll
    for (int r = 0; r < 8; ++r) {
      const int qr = qbase + rb + r;
      const int kc = jt * 16 + col;
      S[(rb + r) * SEQ_T + kc] = (kc <= qr) ? acc[r] * scale : -1e30f;
    }
  }
  __syncthreads();

  // Phase 2: row softmax (2 lanes per row), P = softmax(S) in f16.
  {
    const int row = lane >> 1, hf = lane & 1;
    float* Sr = S + (size_t)row * SEQ_T;
    h16_t* Pr = P + (size_t)row * SEQ_T;
    float m = -1e30f;
    for (int c = hf * 128; c < hf * 128 + 128; ++c) m = fmaxf(m, Sr[c]);
    m = fmaxf(m, __shfl_xor(m, 1, 32));
    float l = 0.f;
    for (int c = hf * 128; c < hf * 128 + 128; ++c) {
      float e = __expf(Sr[c] - m);
      Sr[c] = e;
      l += e;
    }
    l += __shfl_xor(l, 1, 32);
    const float inv = 1.0f / l;
    for (int c = hf * 128; c < hf * 128 + 128; ++c)
      Pr[c] = (h16_t)(Sr[c] * inv);
  }
  __syncthreads();

  // Phase 3: O = P V.  A-frags from LDS (b128 ds loads), B-frags from Vt
  // (K-major -> b128 global loads).  4 column tiles of 16 share each A-frag.
  v8f o0 = {}, o1 = {}, o2 = {}, o3 = {};
  for (int kb = 0; kb < SEQ_T / 32; ++kb) {
    v16h a = load_kfrag(P + kb * 32, SEQ_T, lane);
    const h16_t* Vr = Vtb + kb * 32;
    o0 = wmma_f16(a, load_kfrag(Vr,                       SEQ_T, lane), o0);
    o1 = wmma_f16(a, load_kfrag(Vr + (size_t)16 * SEQ_T,  SEQ_T, lane), o1);
    o2 = wmma_f16(a, load_kfrag(Vr + (size_t)32 * SEQ_T,  SEQ_T, lane), o2);
    o3 = wmma_f16(a, load_kfrag(Vr + (size_t)48 * SEQ_T,  SEQ_T, lane), o3);
  }

  // Store: attn_out[(b*T + t)*C + h*64 + hs]  (bthk layout of the reference)
  const int col = lane & 15, rb = (lane >> 4) * 8;
  h16_t* orow = attn_out + ((size_t)b * SEQ_T + qbase) * N_EMBD + h * HEAD_SZ;
#pragma unroll
  for (int r = 0; r < 8; ++r) {
    const size_t t = (size_t)(rb + r) * N_EMBD;
    orow[t +  0 + col] = (h16_t)o0[r];
    orow[t + 16 + col] = (h16_t)o1[r];
    orow[t + 32 + col] = (h16_t)o2[r];
    orow[t + 48 + col] = (h16_t)o3[r];
  }
}

// ---- Host-side orchestration -----------------------------------------------

static inline char* ws_take(char*& p, size_t bytes) {
  char* r = p;
  p += (bytes + 255) & ~(size_t)255;
  return r;
}

extern "C" void kernel_launch(void* const* d_in, const int* in_sizes, int n_in,
                              void* d_out, int out_size, void* d_ws, size_t ws_size,
                              hipStream_t stream) {
  const float* x     = (const float*)d_in[0];
  const float* Wq    = (const float*)d_in[1];
  const float* Wk    = (const float*)d_in[2];
  const float* Wv    = (const float*)d_in[3];
  const float* Wo    = (const float*)d_in[4];
  const float* bo    = (const float*)d_in[5];
  const float* W1    = (const float*)d_in[6];
  const float* b1    = (const float*)d_in[7];
  const float* W2    = (const float*)d_in[8];
  const float* b2    = (const float*)d_in[9];
  const float* g1    = (const float*)d_in[10];
  const float* beta1 = (const float*)d_in[11];
  const float* g2    = (const float*)d_in[12];
  const float* beta2 = (const float*)d_in[13];
  float* out = (float*)d_out;    // [BT, 384] f32

  // Workspace carve-up (~142 MB total, all f16 intermediates).
  char* p = (char*)d_ws;
  h16_t* WqkvT = (h16_t*)ws_take(p, (size_t)QKV_LD * N_EMBD * 2);   // 1152x384
  h16_t* WoT   = (h16_t*)ws_take(p, (size_t)N_EMBD * N_EMBD * 2);   // 384x384
  h16_t* W1T   = (h16_t*)ws_take(p, (size_t)D_FF * N_EMBD * 2);     // 1536x384
  h16_t* W2T   = (h16_t*)ws_take(p, (size_t)N_EMBD * D_FF * 2);     // 384x1536
  h16_t* h16   = (h16_t*)ws_take(p, (size_t)BT * N_EMBD * 2);       // LN1 out
  h16_t* qkv16 = (h16_t*)ws_take(p, (size_t)BT * QKV_LD * 2);       // QKV
  h16_t* Vt    = (h16_t*)ws_take(p, (size_t)BATCH_B * N_HEADS * HEAD_SZ * SEQ_T * 2);
  h16_t* attn16= (h16_t*)ws_take(p, (size_t)BT * N_EMBD * 2);       // attn out
  h16_t* h216  = (h16_t*)ws_take(p, (size_t)BT * N_EMBD * 2);       // LN2 out
  h16_t* ff116 = (h16_t*)ws_take(p, (size_t)BT * D_FF * 2);         // relu(h2 W1+b1)

  // 1) Pack / convert weights to f16, transposed (K-major).
  pack_qkv_kernel<<<(N_EMBD * QKV_LD + 255) / 256, 256, 0, stream>>>(Wq, Wk, Wv, WqkvT);
  cvt_t_kernel<<<(N_EMBD * N_EMBD + 255) / 256, 256, 0, stream>>>(Wo, WoT, N_EMBD, N_EMBD);
  cvt_t_kernel<<<(N_EMBD * D_FF   + 255) / 256, 256, 0, stream>>>(W1, W1T, N_EMBD, D_FF);
  cvt_t_kernel<<<(D_FF   * N_EMBD + 255) / 256, 256, 0, stream>>>(W2, W2T, D_FF, N_EMBD);

  // 2) h = LN1(x) -> f16
  ln_kernel<<<BT / 8, 256, 0, stream>>>(x, g1, beta1, h16);

  // 3) qkv = h @ Wqkv  [16384 x 1152]
  gemm16_kernel<<<(BT / 256) * (QKV_LD / 64), 256, 0, stream>>>(
      h16, WqkvT, nullptr, nullptr, nullptr, qkv16, BT, QKV_LD, N_EMBD, 0);

  // 3b) Vt = transpose(V)  per (b,h)
  vtrans_kernel<<<BATCH_B * N_HEADS * 4, 256, 0, stream>>>(qkv16, Vt);

  // 4) causal attention -> attn16 [16384 x 384]
  attn_kernel<<<BATCH_B * N_HEADS * 8, 64, 49152, stream>>>(qkv16, Vt, attn16);

  // 5) out = x + attn @ Wo + bo   (f32 residual)
  gemm16_kernel<<<(BT / 256) * (N_EMBD / 64), 256, 0, stream>>>(
      attn16, WoT, bo, x, out, nullptr, BT, N_EMBD, N_EMBD, 0);

  // 6) h2 = LN2(out) -> f16
  ln_kernel<<<BT / 8, 256, 0, stream>>>(out, g2, beta2, h216);

  // 7) ff1 = relu(h2 @ W1 + b1) -> f16 [16384 x 1536]
  gemm16_kernel<<<(BT / 256) * (D_FF / 64), 256, 0, stream>>>(
      h216, W1T, b1, nullptr, nullptr, ff116, BT, D_FF, N_EMBD, 1);

  // 8) out += ff1 @ W2 + b2
  gemm16_kernel<<<(BT / 256) * (N_EMBD / 64), 256, 0, stream>>>(
      ff116, W2T, b2, out, out, nullptr, BT, N_EMBD, D_FF, 0);
}
